// PointNetMSG_73718818669275
// MI455X (gfx1250) — matrix operand order repaired and checked by reading, not compiled, not run on hardware
//
#include <hip/hip_runtime.h>

// ---------------------------------------------------------------------------
// CDNA5 (gfx1250, wave32) implementation of the PointNet-MSG reference.
// Heavy dense math runs on v_wmma_f32_16x16x32_bf16 (bf16 in, f32 accum).
// ---------------------------------------------------------------------------

typedef __attribute__((ext_vector_type(16))) __bf16 v16bf;
typedef __attribute__((ext_vector_type(8)))  float  v8f;

constexpr int   B_   = 8;
constexpr int   N_   = 8192;
constexpr int   NC_  = 1024;
constexpr int   K0_  = 32;
constexpr int   K1_  = 64;
constexpr float R0SQ = 100.0f;   // 10^2
constexpr float R1SQ = 196.0f;   // 14^2
constexpr float BN_EPS_ = 1e-5f;

__device__ __forceinline__ v8f wmma_bf16(v16bf a, v16bf b, v8f c) {
  return __builtin_amdgcn_wmma_f32_16x16x32_bf16(false, a, false, b, (short)0, c, false, false);
}

__device__ __forceinline__ v8f zero8() {
  v8f z = {0.f,0.f,0.f,0.f,0.f,0.f,0.f,0.f};
  return z;
}

// A fragment (16x32 bf16) from row-major f32 weights W[m][k], leading dim ldw (mult of 4).
// Layout: lane<16 -> row m0+lane, K = {k0..k0+7, k0+16..k0+23}; lane>=16 -> +8 shift.
// Two contiguous 8-float runs -> 4x global_load_b128.
__device__ __forceinline__ v16bf load_a_frag_f32(const float* W, int ldw, int m0, int k0) {
  const int lane = threadIdx.x & 31;
  const float* p = W + (size_t)(m0 + (lane & 15)) * ldw + k0 + ((lane >> 4) << 3);
  const float4* p4 = (const float4*)p;
  float4 a0 = p4[0], a1 = p4[1];     // p[0..7]
  float4 b0 = p4[4], b1 = p4[5];     // p[16..23]
  v16bf a;
  a[0]  = (__bf16)a0.x; a[1]  = (__bf16)a0.y; a[2]  = (__bf16)a0.z; a[3]  = (__bf16)a0.w;
  a[4]  = (__bf16)a1.x; a[5]  = (__bf16)a1.y; a[6]  = (__bf16)a1.z; a[7]  = (__bf16)a1.w;
  a[8]  = (__bf16)b0.x; a[9]  = (__bf16)b0.y; a[10] = (__bf16)b0.z; a[11] = (__bf16)b0.w;
  a[12] = (__bf16)b1.x; a[13] = (__bf16)b1.y; a[14] = (__bf16)b1.z; a[15] = (__bf16)b1.w;
  return a;
}

// B fragment (32x16 bf16) from column-major bf16 LDS tile base[col][ldc], cols c0..c0+15, rows k0..k0+31.
__device__ __forceinline__ v16bf load_b_frag_lds(const __bf16* base, int ldc, int c0, int k0) {
  const int lane = threadIdx.x & 31;
  const __bf16* p = base + (size_t)(c0 + (lane & 15)) * ldc + k0 + ((lane >> 4) << 3);
  v16bf b;
#pragma unroll
  for (int j = 0; j < 8; ++j) { b[j] = p[j]; b[j + 8] = p[16 + j]; }
  return b;
}

// Squared distance between two 64-float rows (both 16B aligned).
__device__ __forceinline__ float dist64(const float* __restrict__ a, const float* __restrict__ b) {
  const float4* a4 = (const float4*)a;
  const float4* b4 = (const float4*)b;
  float d = 0.f;
#pragma unroll
  for (int q = 0; q < 16; ++q) {
    float4 x = a4[q], y = b4[q];
    float t0 = x.x - y.x, t1 = x.y - y.y, t2 = x.z - y.z, t3 = x.w - y.w;
    d += t0 * t0 + t1 * t1 + t2 * t2 + t3 * t3;
  }
  return d;
}

// ---------------------------------------------------------------------------
// 1) support^T : supportT[b][n][c] (c<3: xyz, else point_features)
// ---------------------------------------------------------------------------
__global__ __launch_bounds__(256) void k_build(const float* __restrict__ xyz,
                                               const float* __restrict__ pf,
                                               float* __restrict__ supportT) {
  int idx = blockIdx.x * 256 + threadIdx.x;            // over B*N*64 = 2^22
  int c = idx & 63, n = (idx >> 6) & 8191, b = idx >> 19;
  float v = (c < 3) ? xyz[((size_t)(b * 3 + c)) * N_ + n]
                    : pf [((size_t)(b * 61 + (c - 3))) * N_ + n];
  supportT[idx] = v;
}

// ---------------------------------------------------------------------------
// 2) Farthest point sampling: 1 workgroup (32 waves) per batch, sequential NC steps.
//    Matches jnp scan semantics: cents[i] = incoming "far", argmax ties -> lowest index.
// ---------------------------------------------------------------------------
__global__ __launch_bounds__(1024) void k_fps(const float* __restrict__ supportT,
                                              int* __restrict__ cents) {
  __shared__ alignas(16) float scen[64];
  __shared__ float svals[1024];
  __shared__ int   sidx [1024];
  const int b = blockIdx.x, tid = threadIdx.x;
  const float* sp = supportT + (size_t)b * N_ * 64;
  float dist[8];
#pragma unroll
  for (int j = 0; j < 8; ++j) dist[j] = 1e10f;
  int far = 0;
  for (int i = 0; i < NC_; ++i) {
    if (tid == 0) cents[b * NC_ + i] = far;
    if (tid < 64) scen[tid] = sp[(size_t)far * 64 + tid];
    __syncthreads();
    float bestv = -1.0f; int besti = 0;
#pragma unroll
    for (int j = 0; j < 8; ++j) {
      const int p = tid + (j << 10);
      const float* pr = sp + (size_t)p * 64;
      __builtin_prefetch(pr + 64 * 1024, 0, 1);        // global_prefetch_b8 for next chunk
      float d = dist64(pr, scen);
      float nd = fminf(dist[j], d);
      dist[j] = nd;
      if (nd > bestv) { bestv = nd; besti = p; }       // ascending p => first max kept
    }
    svals[tid] = bestv; sidx[tid] = besti;
    __syncthreads();
    for (int s = 512; s > 0; s >>= 1) {
      if (tid < s) {
        float v2 = svals[tid + s]; int i2 = sidx[tid + s];
        if (v2 > svals[tid] || (v2 == svals[tid] && i2 < sidx[tid])) { svals[tid] = v2; sidx[tid] = i2; }
      }
      __syncthreads();
    }
    far = sidx[0];
    __syncthreads();
  }
}

// ---------------------------------------------------------------------------
// 3) Gather centroids (point-major) + emit new_xyz into d_out.
// ---------------------------------------------------------------------------
__global__ __launch_bounds__(256) void k_gather(const float* __restrict__ supportT,
                                                const int* __restrict__ cents,
                                                float* __restrict__ centroidsT,
                                                float* __restrict__ newxyz) {
  int idx = blockIdx.x * 256 + threadIdx.x;            // over B*NC*64 = 2^19
  int c = idx & 63, i = (idx >> 6) & 1023, b = idx >> 16;
  int src = cents[b * NC_ + i];
  float v = supportT[((size_t)(b * N_ + src)) * 64 + c];
  centroidsT[idx] = v;
  if (c < 3) newxyz[((size_t)(b * 3 + c)) * NC_ + i] = v;
}

// ---------------------------------------------------------------------------
// 4) Ball grouping, both radii in one pass. One wave32 per centroid.
//    Selection = first-K in-ball indices in increasing order (== reference top_k trick),
//    pad with first in-ball index, mask=1 on padded slots.
// ---------------------------------------------------------------------------
__global__ __launch_bounds__(128) void k_ballgroup(const float* __restrict__ supportT,
                                                   const float* __restrict__ centroidsT,
                                                   int* __restrict__ gidx0, int* __restrict__ mask0,
                                                   int* __restrict__ gidx1, int* __restrict__ mask1) {
  __shared__ alignas(16) float scen[4][64];
  const int lane = threadIdx.x & 31, wid = threadIdx.x >> 5;
  const int wave = blockIdx.x * 4 + wid;               // grid = B*NC/4 exactly
  const int b = wave >> 10;
  scen[wid][lane]      = centroidsT[(size_t)wave * 64 + lane];
  scen[wid][lane + 32] = centroidsT[(size_t)wave * 64 + 32 + lane];
  __syncthreads();
  const float* cen = scen[wid];
  const size_t g0 = (size_t)wave * K0_;
  const size_t g1 = (size_t)wave * K1_;
  int cnt0 = 0, cnt1 = 0, first0 = 0, first1 = 0;
  const unsigned ltmask = (1u << lane) - 1u;
  for (int p0 = 0; p0 < N_; p0 += 32) {
    const int p = p0 + lane;
    const float* pr = supportT + ((size_t)(b * N_ + p)) * 64;
    float d = dist64(pr, cen);
    const bool in0 = d <= R0SQ, in1 = d <= R1SQ;
    unsigned m0 = (unsigned)__ballot(in0);
    unsigned m1 = (unsigned)__ballot(in1);
    if (cnt0 < K0_) {
      if (cnt0 == 0 && m0) first0 = p0 + __ffs((int)m0) - 1;
      int pos = cnt0 + __popc(m0 & ltmask);
      if (in0 && pos < K0_) gidx0[g0 + pos] = p;
      cnt0 = min(K0_, cnt0 + __popc(m0));
    }
    if (cnt1 < K1_) {
      if (cnt1 == 0 && m1) first1 = p0 + __ffs((int)m1) - 1;
      int pos = cnt1 + __popc(m1 & ltmask);
      if (in1 && pos < K1_) gidx1[g1 + pos] = p;
      cnt1 = min(K1_, cnt1 + __popc(m1));
    }
    if (cnt0 >= K0_ && cnt1 >= K1_) break;
  }
  for (int t = lane; t < K0_; t += 32) {
    if (t >= cnt0) gidx0[g0 + t] = first0;
    mask0[g0 + t] = (t < cnt0) ? 0 : 1;
  }
  for (int t = lane; t < K1_; t += 32) {
    if (t >= cnt1) gidx1[g1 + t] = first1;
    mask1[g1 + t] = (t < cnt1) ? 0 : 1;
  }
}

// ---------------------------------------------------------------------------
// 5) Fused multi-head ball attention for a 16-centroid tile.
//    Refactor: scores = (Wk[h]^T q[h])^T G ; out = Wv[h] (G w^T). All dense steps = WMMA.
// ---------------------------------------------------------------------------
template<int K>
__global__ __launch_bounds__(128) void k_attn(const float* __restrict__ supportT,
                                              const float* __restrict__ centroidsT,
                                              const int* __restrict__ gidx,
                                              const int* __restrict__ gmask,
                                              const float* __restrict__ Wq,
                                              const float* __restrict__ Wk,
                                              const float* __restrict__ Wv,
                                              const float* __restrict__ Wo,
                                              float* __restrict__ att_out) {
  __shared__ alignas(16) float  sCf [16][64];    // centroid cols f32
  __shared__ alignas(16) __bf16 sCb [16][64];    // centroid cols bf16 (B operand)
  __shared__ alignas(16) __bf16 sQb [16][128];   // q col-major bf16
  __shared__ alignas(16) float  sQK [32][64];    // (Wk^T q), row = h*16+n
  __shared__ alignas(16) float  sW  [2][16][K];  // scores / softmax weights
  __shared__ alignas(16) int    sGidx[16][K];
  __shared__ alignas(16) int    sMask[16][K];
  __shared__ alignas(16) __bf16 sGwB[32][64];    // G * w^T, col = h*16+n
  __shared__ alignas(16) __bf16 sOutB[16][128];  // attention "out", col-major

  const int tid = threadIdx.x, lane = tid & 31, wid = tid >> 5;
  const int b  = blockIdx.x >> 6;                // NC/16 = 64 tiles
  const int n0 = (blockIdx.x & 63) << 4;

  for (int idx = tid; idx < 16 * 64; idx += 128) {
    int n = idx >> 6, c = idx & 63;
    float v = centroidsT[((size_t)(b * NC_ + n0 + n)) * 64 + c];
    sCf[n][c] = v; sCb[n][c] = (__bf16)v;
  }
  for (int idx = tid; idx < 16 * K; idx += 128) {
    int n = idx / K, k = idx % K;
    size_t off = ((size_t)(b * NC_ + n0 + n)) * K + k;
    sGidx[n][k] = gidx[off];
    sMask[n][k] = gmask[off];
  }
  __syncthreads();

  // q = Wq @ C  (128 x 16), 2 M-tiles per wave
  for (int mt = wid; mt < 8; mt += 4) {
    v8f acc = zero8();
#pragma unroll
    for (int ks = 0; ks < 64; ks += 32) {
      v16bf a  = load_a_frag_f32(Wq, 64, mt * 16, ks);
      v16bf bb = load_b_frag_lds(&sCb[0][0], 64, 0, ks);
      acc = wmma_bf16(a, bb, acc);
    }
    int col = lane & 15, mb = mt * 16 + ((lane >> 4) << 3);
#pragma unroll
    for (int g = 0; g < 8; ++g) sQb[col][mb + g] = (__bf16)acc[g];
  }
  __syncthreads();

  // qk[h] = Wk[h]^T @ q[h]  (64 x 16 per head); Wk^T access is column-strided (small)
  for (int job = wid; job < 8; job += 4) {
    int h = job >> 2, m0 = (job & 3) * 16;
    v8f acc = zero8();
#pragma unroll
    for (int ks = 0; ks < 64; ks += 32) {
      v16bf a;
      int m  = m0 + (lane & 15);
      int kb = h * 64 + ks + ((lane >> 4) << 3);
#pragma unroll
      for (int j = 0; j < 8; ++j) a[j]     = (__bf16)Wk[(size_t)(kb + j) * 64 + m];
#pragma unroll
      for (int j = 0; j < 8; ++j) a[j + 8] = (__bf16)Wk[(size_t)(kb + 16 + j) * 64 + m];
      v16bf bb = load_b_frag_lds(&sQb[0][0], 128, 0, h * 64 + ks);
      acc = wmma_bf16(a, bb, acc);
    }
    int col = lane & 15, mb = m0 + ((lane >> 4) << 3);
#pragma unroll
    for (int g = 0; g < 8; ++g) sQK[h * 16 + col][mb + g] = acc[g];
  }
  __syncthreads();

  // scores (both heads per neighbor column, gather straight from L2-resident supportT)
  for (int idx = tid; idx < 16 * K; idx += 128) {
    int n = idx / K, k = idx % K;
    const float4* pp  = (const float4*)(supportT + ((size_t)(b * N_ + sGidx[n][k])) * 64);
    const float4* cen = (const float4*)sCf[n];
    const float4* q0  = (const float4*)sQK[n];
    const float4* q1  = (const float4*)sQK[16 + n];
    float s0 = 0.f, s1 = 0.f;
#pragma unroll
    for (int q = 0; q < 16; ++q) {
      float4 gv = pp[q], cv = cen[q], a0 = q0[q], a1 = q1[q];
      float g0 = gv.x - cv.x, g1 = gv.y - cv.y, g2 = gv.z - cv.z, g3 = gv.w - cv.w;
      s0 += g0 * a0.x + g1 * a0.y + g2 * a0.z + g3 * a0.w;
      s1 += g0 * a1.x + g1 * a1.y + g2 * a1.z + g3 * a1.w;
    }
    bool mk = sMask[n][k] != 0;
    sW[0][n][k] = mk ? -1e9f : s0 * 0.125f;   // / sqrt(64)
    sW[1][n][k] = mk ? -1e9f : s1 * 0.125f;
  }
  __syncthreads();

  // softmax per (h, n)
  if (tid < 32) {
    int h = tid >> 4, n = tid & 15;
    float mx = -3.4e38f;
    for (int k = 0; k < K; ++k) mx = fmaxf(mx, sW[h][n][k]);
    float sum = 0.f;
    for (int k = 0; k < K; ++k) { float e = __expf(sW[h][n][k] - mx); sW[h][n][k] = e; sum += e; }
    float inv = 1.f / sum;
    for (int k = 0; k < K; ++k) sW[h][n][k] *= inv;
  }
  __syncthreads();

  // Gw[c][(h,n)] = sum_k w * (support[gidx]-cen)
  {
    int hn = tid >> 2, h = hn >> 4, n = hn & 15, cb = (tid & 3) << 4;
    float acc[16];
#pragma unroll
    for (int j = 0; j < 16; ++j) acc[j] = 0.f;
    for (int k = 0; k < K; ++k) {
      float w = sW[h][n][k];
      const float4* pp  = (const float4*)(supportT + ((size_t)(b * N_ + sGidx[n][k])) * 64 + cb);
      const float4* cen = (const float4*)(&sCf[n][cb]);
#pragma unroll
      for (int q = 0; q < 4; ++q) {
        float4 gv = pp[q], cv = cen[q];
        acc[q * 4 + 0] += w * (gv.x - cv.x);
        acc[q * 4 + 1] += w * (gv.y - cv.y);
        acc[q * 4 + 2] += w * (gv.z - cv.z);
        acc[q * 4 + 3] += w * (gv.w - cv.w);
      }
    }
#pragma unroll
    for (int j = 0; j < 16; ++j) sGwB[hn][cb + j] = (__bf16)acc[j];
  }
  __syncthreads();

  // out[h] = Wv[h] @ Gw_h  (64 x 16 per head)
  for (int job = wid; job < 8; job += 4) {
    int h = job >> 2, m0 = (job & 3) * 16;
    v8f acc = zero8();
#pragma unroll
    for (int ks = 0; ks < 64; ks += 32) {
      v16bf a  = load_a_frag_f32(Wv + (size_t)h * 64 * 64, 64, m0, ks);
      v16bf bb = load_b_frag_lds(&sGwB[0][0], 64, h * 16, ks);
      acc = wmma_bf16(a, bb, acc);
    }
    int col = lane & 15, mb = h * 64 + m0 + ((lane >> 4) << 3);
#pragma unroll
    for (int g = 0; g < 8; ++g) sOutB[col][mb + g] = (__bf16)acc[g];
  }
  __syncthreads();

  // f = Wo @ out + centroid (residual)
  {
    int m0 = wid * 16;
    v8f acc = zero8();
#pragma unroll
    for (int ks = 0; ks < 128; ks += 32) {
      v16bf a  = load_a_frag_f32(Wo, 128, m0, ks);
      v16bf bb = load_b_frag_lds(&sOutB[0][0], 128, 0, ks);
      acc = wmma_bf16(a, bb, acc);
    }
    int col = lane & 15, mb = m0 + ((lane >> 4) << 3);
#pragma unroll
    for (int g = 0; g < 8; ++g) {
      int row = mb + g;
      att_out[((size_t)(b * 64 + row)) * NC_ + n0 + col] = acc[g] + sCf[col][row];
    }
  }
}

// ---------------------------------------------------------------------------
// 6) 1x1 conv (WMMA) + bias, optional preceding BN+LeakyReLU, BN stat atomics.
// ---------------------------------------------------------------------------
template<int CO, bool PRE>
__global__ __launch_bounds__(128) void k_conv_bn(const float* __restrict__ in,
                                                 const float* __restrict__ W,
                                                 const float* __restrict__ bias,
                                                 const float* __restrict__ pre_a,
                                                 const float* __restrict__ pre_b,
                                                 float* __restrict__ out,
                                                 float* __restrict__ sums,
                                                 float* __restrict__ sumsq) {
  __shared__ alignas(16) __bf16 sXb[16][64];
  __shared__ alignas(16) float  sO [CO][16];
  const int tid = threadIdx.x, lane = tid & 31, wid = tid >> 5;
  const int b  = blockIdx.x >> 6;
  const int n0 = (blockIdx.x & 63) << 4;
  // point index fastest across threads -> coalesced global reads
  for (int idx = tid; idx < 16 * 64; idx += 128) {
    int n = idx & 15, c = idx >> 4;
    float v = in[((size_t)(b * 64 + c)) * NC_ + n0 + n];
    if (PRE) { v = pre_a[c] * v + pre_b[c]; v = (v >= 0.f) ? v : 0.02f * v; }
    sXb[n][c] = (__bf16)v;
  }
  __syncthreads();
  for (int mt = wid; mt < CO / 16; mt += 4) {
    v8f acc = zero8();
#pragma unroll
    for (int ks = 0; ks < 64; ks += 32) {
      v16bf a  = load_a_frag_f32(W, 64, mt * 16, ks);
      v16bf bb = load_b_frag_lds(&sXb[0][0], 64, 0, ks);
      acc = wmma_bf16(a, bb, acc);
    }
    int col = lane & 15, mb = mt * 16 + ((lane >> 4) << 3);
#pragma unroll
    for (int g = 0; g < 8; ++g) {
      int row = mb + g;
      float v = acc[g] + bias[row];
      sO[row][col] = v;
      out[((size_t)(b * CO + row)) * NC_ + n0 + col] = v;
    }
  }
  __syncthreads();
  if (tid < CO) {
    float s = 0.f, s2 = 0.f;
#pragma unroll
    for (int n = 0; n < 16; ++n) { float v = sO[tid][n]; s += v; s2 += v * v; }
    atomicAdd(&sums[tid], s);
    atomicAdd(&sumsq[tid], s2);
  }
}

__global__ void k_bn_fin(const float* __restrict__ sums, const float* __restrict__ sumsq,
                         const float* __restrict__ gamma, const float* __restrict__ beta,
                         float* __restrict__ a, float* __restrict__ bsh, int CO) {
  int c = threadIdx.x;
  if (c >= CO) return;
  const float cnt = (float)(B_ * NC_);
  float mu  = sums[c] / cnt;
  float var = sumsq[c] / cnt - mu * mu;      // biased var, matches jnp.var
  float aa  = gamma[c] * rsqrtf(var + BN_EPS_);
  a[c]   = aa;
  bsh[c] = beta[c] - aa * mu;
}

// ---------------------------------------------------------------------------
// 7) Time embedding MLP: te = silu(t @ t1^T + b1) @ t2^T + b2. One WG per batch.
// ---------------------------------------------------------------------------
__global__ __launch_bounds__(128) void k_te(const float* __restrict__ t,
                                            const float* __restrict__ t1w, const float* __restrict__ t1b,
                                            const float* __restrict__ t2w, const float* __restrict__ t2b,
                                            float* __restrict__ te) {
  __shared__ float st[128];
  __shared__ float sh[512];
  const int b = blockIdx.x, tid = threadIdx.x;
  st[tid] = t[b * 128 + tid];
  __syncthreads();
  for (int o = tid; o < 512; o += 128) {
    float s = t1b[o];
    const float* w = t1w + (size_t)o * 128;
#pragma unroll 8
    for (int c = 0; c < 128; ++c) s += st[c] * w[c];
    sh[o] = s / (1.f + __expf(-s));          // silu
  }
  __syncthreads();
  float s = t2b[tid];
  const float* w = t2w + (size_t)tid * 512;
#pragma unroll 8
  for (int j = 0; j < 512; ++j) s += sh[j] * w[j];
  te[b * 128 + tid] = s;
}

__global__ __launch_bounds__(256) void k_final(const float* __restrict__ h2,
                                               const float* __restrict__ a, const float* __restrict__ bsh,
                                               const float* __restrict__ te,
                                               float* __restrict__ outfeat, int scale_off) {
  size_t idx = (size_t)blockIdx.x * 256 + threadIdx.x;   // over B*128*NC
  int n = idx & 1023, ch = (idx >> 10) & 127, b = (int)(idx >> 17);
  float v = a[ch] * h2[idx] + bsh[ch] + te[b * 128 + ch];
  outfeat[((size_t)(b * 256 + scale_off + ch)) * NC_ + n] = v;
}

// ---------------------------------------------------------------------------
// Host side
// ---------------------------------------------------------------------------
struct SParams {
  const float *bn_b0, *bn_b1, *bn_g0, *bn_g1;
  const float *conv_b0, *conv_b1, *conv_w0, *conv_w1;
  const float *t1_b, *t1_w, *t2_b, *t2_w;
  const float *wk, *wo, *wq, *wv;
};

// Nested param dicts assumed flattened in JAX sorted-key order:
// bn_b[0,1], bn_g[0,1], conv_b[0,1], conv_w[0,1], t1_b, t1_w, t2_b, t2_w, wk, wo, wq, wv
static SParams unpack(void* const* d_in, int base) {
  SParams p;
  p.bn_b0   = (const float*)d_in[base + 0];
  p.bn_b1   = (const float*)d_in[base + 1];
  p.bn_g0   = (const float*)d_in[base + 2];
  p.bn_g1   = (const float*)d_in[base + 3];
  p.conv_b0 = (const float*)d_in[base + 4];
  p.conv_b1 = (const float*)d_in[base + 5];
  p.conv_w0 = (const float*)d_in[base + 6];
  p.conv_w1 = (const float*)d_in[base + 7];
  p.t1_b    = (const float*)d_in[base + 8];
  p.t1_w    = (const float*)d_in[base + 9];
  p.t2_b    = (const float*)d_in[base + 10];
  p.t2_w    = (const float*)d_in[base + 11];
  p.wk      = (const float*)d_in[base + 12];
  p.wo      = (const float*)d_in[base + 13];
  p.wq      = (const float*)d_in[base + 14];
  p.wv      = (const float*)d_in[base + 15];
  return p;
}

extern "C" void kernel_launch(void* const* d_in, const int* in_sizes, int n_in,
                              void* d_out, int out_size, void* d_ws, size_t ws_size,
                              hipStream_t stream) {
  (void)in_sizes; (void)n_in; (void)out_size; (void)ws_size;
  const float* xyz      = (const float*)d_in[0];
  const float* time_emb = (const float*)d_in[1];
  const float* pf       = (const float*)d_in[2];
  SParams P0 = unpack(d_in, 3);
  SParams P1 = unpack(d_in, 19);

  char* w = (char*)d_ws;
  auto alloc = [&](size_t bytes) -> char* {
    char* r = w; w += (bytes + 255) & ~(size_t)255; return r;
  };
  float* supportT   = (float*)alloc((size_t)B_ * N_ * 64 * 4);
  int*   cents      = (int*)  alloc((size_t)B_ * NC_ * 4);
  float* centroidsT = (float*)alloc((size_t)B_ * NC_ * 64 * 4);
  int*   gidx0      = (int*)  alloc((size_t)B_ * NC_ * K0_ * 4);
  int*   mask0      = (int*)  alloc((size_t)B_ * NC_ * K0_ * 4);
  int*   gidx1      = (int*)  alloc((size_t)B_ * NC_ * K1_ * 4);
  int*   mask1      = (int*)  alloc((size_t)B_ * NC_ * K1_ * 4);
  float* att_out    = (float*)alloc((size_t)B_ * 64 * NC_ * 4);
  float* h1         = (float*)alloc((size_t)B_ * 64 * NC_ * 4);
  float* h2         = (float*)alloc((size_t)B_ * 128 * NC_ * 4);
  float* stats      = (float*)alloc(256 * 4);      // sums[0:128] | sumsq[128:256]
  float* bn_a       = (float*)alloc(128 * 4);
  float* bn_s       = (float*)alloc(128 * 4);
  float* te         = (float*)alloc((size_t)B_ * 128 * 4);

  float* outF   = (float*)d_out;
  float* newxyz = outF;
  float* feats  = outF + (size_t)B_ * 3 * NC_;

  k_build<<<(B_ * N_ * 64) / 256, 256, 0, stream>>>(xyz, pf, supportT);
  k_fps<<<B_, 1024, 0, stream>>>(supportT, cents);
  k_gather<<<(B_ * NC_ * 64) / 256, 256, 0, stream>>>(supportT, cents, centroidsT, newxyz);
  k_ballgroup<<<(B_ * NC_) / 4, 128, 0, stream>>>(supportT, centroidsT, gidx0, mask0, gidx1, mask1);

  const int tiles = B_ * (NC_ / 16);   // 512
  for (int s = 0; s < 2; ++s) {
    const SParams& P = (s == 0) ? P0 : P1;
    if (s == 0)
      k_attn<K0_><<<tiles, 128, 0, stream>>>(supportT, centroidsT, gidx0, mask0,
                                             P.wq, P.wk, P.wv, P.wo, att_out);
    else
      k_attn<K1_><<<tiles, 128, 0, stream>>>(supportT, centroidsT, gidx1, mask1,
                                             P.wq, P.wk, P.wv, P.wo, att_out);

    hipMemsetAsync(stats, 0, 256 * sizeof(float), stream);
    k_conv_bn<64, false><<<tiles, 128, 0, stream>>>(att_out, P.conv_w0, P.conv_b0,
                                                    nullptr, nullptr, h1,
                                                    stats, stats + 128);
    k_bn_fin<<<1, 128, 0, stream>>>(stats, stats + 128, P.bn_g0, P.bn_b0, bn_a, bn_s, 64);

    hipMemsetAsync(stats, 0, 256 * sizeof(float), stream);
    k_conv_bn<128, true><<<tiles, 128, 0, stream>>>(h1, P.conv_w1, P.conv_b1,
                                                    bn_a, bn_s, h2,
                                                    stats, stats + 128);
    k_bn_fin<<<1, 128, 0, stream>>>(stats, stats + 128, P.bn_g1, P.bn_b1, bn_a, bn_s, 128);

    k_te<<<B_, 128, 0, stream>>>(time_emb, P.t1_w, P.t1_b, P.t2_w, P.t2_b, te);
    k_final<<<(B_ * 128 * NC_) / 256, 256, 0, stream>>>(h2, bn_a, bn_s, te, feats, s * 128);
  }
}
